// Embedding_SimCLR_Loss_73340861546690
// MI455X (gfx1250) — matrix operand order repaired and checked by reading, not compiled, not run on hardware
//
#include <hip/hip_runtime.h>
#include <hip/hip_bf16.h>
#include <math.h>

typedef __attribute__((ext_vector_type(16))) _Float16 v16h;
typedef __attribute__((ext_vector_type(8)))  float    v8f;

constexpr int B = 8, C = 128, H = 160, W = 160;
constexpr int P = H * W;               // 25600 pixels per image
constexpr int K = 20;
constexpr int N2 = 2 * K;              // 40 rows in the SimCLR sim matrix
constexpr float ALPHA    = 0.02f;
constexpr float INV_TEMP = 2.0f;       // 1 / 0.5

// ---------------- workspace layout (floats) ----------------
constexpr size_t WS_SUMS   = 0;                         // B*C*K
constexpr size_t WS_COUNTS = WS_SUMS   + (size_t)B*C*K; // B*K
constexpr size_t WS_MEANS  = WS_COUNTS + (size_t)B*K;   // B*C*K
constexpr size_t WS_ACC    = WS_MEANS  + (size_t)B*C*K; // [0]=intra sum, [1]=simclr mean
constexpr size_t WS_TOTAL  = WS_ACC + 2;

// ---------------- kernel 0: zero workspace ----------------
__global__ void k_zero(float* __restrict__ ws, int n) {
    int i = blockIdx.x * blockDim.x + threadIdx.x;
    if (i < n) ws[i] = 0.0f;
}

// ---------------- kernel 1: segmented sums + counts ----------------
constexpr int T1     = 256;
constexpr int PPB    = 512;            // pixels per block
constexpr int CHUNKS = P / PPB;        // 50

__global__ __launch_bounds__(T1) void k_seg_sums(
        const float* __restrict__ x, const int* __restrict__ mask,
        float* __restrict__ sums, float* __restrict__ counts) {
    // padded stride 129 -> bank index (k + c) mod 64: spreads the per-channel
    // atomic traffic that a stride-128 layout would serialize onto one bank.
    __shared__ float ls[K * 129];
    __shared__ int   lc[K];
    const int t    = threadIdx.x;
    const int b    = blockIdx.x / CHUNKS;
    const int base = (blockIdx.x % CHUNKS) * PPB;

    for (int i = t; i < K * 129; i += T1) ls[i] = 0.0f;
    if (t < K) lc[t] = 0;
    __syncthreads();

    const int p0 = base + t;
    const int p1 = base + T1 + t;
    const int k0 = mask[b * P + p0];
    const int k1 = mask[b * P + p1];
    atomicAdd(&lc[k0], 1);
    atomicAdd(&lc[k1], 1);

    const float* xb = x + (size_t)b * C * P;
    for (int c = 0; c < C; ++c) {
        float v0 = xb[(size_t)c * P + p0];   // coalesced across lanes
        float v1 = xb[(size_t)c * P + p1];
        atomicAdd(&ls[k0 * 129 + c], v0);    // ds_add_f32
        atomicAdd(&ls[k1 * 129 + c], v1);
    }
    __syncthreads();

    for (int i = t; i < K * C; i += T1) {
        int k = i / C, c = i - k * C;
        atomicAdd(&sums[((size_t)b * C + c) * K + k], ls[k * 129 + c]);
    }
    if (t < K) atomicAdd(&counts[b * K + t], (float)lc[t]);
}

// ---------------- kernel 2: normalized cluster means ----------------
__global__ __launch_bounds__(128) void k_means(
        const float* __restrict__ sums, const float* __restrict__ counts,
        float* __restrict__ means) {
    __shared__ float red[128];
    const int b = blockIdx.x / K;
    const int k = blockIdx.x % K;
    const int c = threadIdx.x;
    float val = sums[((size_t)b * C + c) * K + k] / (counts[b * K + k] + 1e-10f);
    red[c] = val * val;
    __syncthreads();
    for (int s = 64; s > 0; s >>= 1) {
        if (c < s) red[c] += red[c + s];
        __syncthreads();
    }
    float nrm = sqrtf(red[0]);
    means[((size_t)b * C + c) * K + k] = val / (nrm + 1e-12f);
}

// ---------------- kernel 3: intra loss (second pass over x) ----------------
constexpr int T3      = 256;
constexpr int PPB3    = 1024;
constexpr int CHUNKS3 = P / PPB3;      // 25

__global__ __launch_bounds__(T3) void k_intra(
        const float* __restrict__ x, const int* __restrict__ mask,
        const float* __restrict__ means, float* __restrict__ acc_intra) {
    __shared__ float lm[K * 129];
    __shared__ float red[T3];
    const int t    = threadIdx.x;
    const int b    = blockIdx.x / CHUNKS3;
    const int base = (blockIdx.x % CHUNKS3) * PPB3;

    for (int i = t; i < K * C; i += T3) {       // means[b] -> transposed, padded LDS
        int c = i / K, k = i - c * K;
        lm[k * 129 + c] = means[(size_t)b * C * K + i];
    }
    __syncthreads();

    const float* xb = x + (size_t)b * C * P;
    float local = 0.0f;
    for (int j = 0; j < PPB3 / T3; ++j) {
        int p = base + j * T3 + t;
        int k = mask[b * P + p];
        float dot = 0.0f;
        for (int c = 0; c < C; ++c)
            dot += xb[(size_t)c * P + p] * lm[k * 129 + c];
        float intra = 0.5f * (1.0f - dot);
        if (intra - ALPHA > 0.0f) local += intra * intra;
    }
    red[t] = local;
    __syncthreads();
    for (int s = T3 / 2; s > 0; s >>= 1) {
        if (t < s) red[t] += red[t + s];
        __syncthreads();
    }
    if (t == 0) atomicAdd(acc_intra, red[0]);
}

// ---------------- kernel 4: SimCLR via WMMA ----------------
// One block per pair; 9 waves compute the 3x3 grid of 16x16 tiles of
// sim = zn (48x128, rows 40..47 zero) @ zn^T using v_wmma_f32_16x16x32_f16.
constexpr int ZS = 129;                // padded zn row stride (floats)

__global__ __launch_bounds__(288) void k_simclr(
        const float* __restrict__ means, float* __restrict__ acc_simclr) {
    __shared__ float zn[48 * ZS];
    __shared__ float sim[N2 * 41];
    __shared__ float lrow[N2];
    const int t    = threadIdx.x;
    const int pair = blockIdx.x;       // 0..3

    // Build zn rows: row r<K from batch `pair` cluster r, row r>=K from batch
    // pair+4 cluster r-K. Reference re-normalizes the already-unit means with
    // +1e-8 in the denominator; replicate that exactly.
    if (t < 48) {
        if (t < N2) {
            const int src_b = (t < K) ? pair : pair + 4;
            const int src_k = (t < K) ? t : t - K;
            const float* mb = means + (size_t)src_b * C * K + src_k;
            float ss = 0.0f;
            for (int c = 0; c < C; ++c) {
                float v = mb[(size_t)c * K];
                zn[t * ZS + c] = v;
                ss += v * v;
            }
            float inv = 1.0f / (sqrtf(ss) + 1e-8f);
            for (int c = 0; c < C; ++c) zn[t * ZS + c] *= inv;
        } else {
            for (int c = 0; c < C; ++c) zn[t * ZS + c] = 0.0f;  // pad rows
        }
    }
    __syncthreads();

    const int wave = t >> 5;
    const int l    = t & 31;
    const int tM   = wave / 3, tN = wave % 3;
    const int khi  = l >> 4;                   // lane-half selects K sub-block
    const int mrow = tM * 16 + (l & 15);       // A: lanes = M
    const int kBl  = (l & 15) + (khi << 4);    // unused; B uses lanes = N

    v8f accv = {};
    for (int kb = 0; kb < 4; ++kb) {
        v16h a, bf;
#pragma unroll
        for (int i = 0; i < 8; ++i) {          // A 16x32 layout (ISA 7.12.2)
            int kk = kb * 32 + ((i >> 2) << 4) + ((i & 3) << 1) + (khi << 3);
            a[2 * i]     = (_Float16)zn[mrow * ZS + kk];
            a[2 * i + 1] = (_Float16)zn[mrow * ZS + kk + 1];
        }
        int n = tN * 16 + (l & 15);
#pragma unroll
        for (int j = 0; j < 8; ++j) {          // B 32x16: lanes = N, elements = K (same order as A)
            int kk2 = kb * 32 + ((j >> 2) << 4) + ((j & 3) << 1) + (khi << 3);
            bf[2 * j]     = (_Float16)zn[n * ZS + kk2];
            bf[2 * j + 1] = (_Float16)zn[n * ZS + kk2 + 1];
        }
        accv = __builtin_amdgcn_wmma_f32_16x16x32_f16(
            false, a, false, bf, (short)0, accv, false, false);
    }
#pragma unroll
    for (int v = 0; v < 8; ++v) {              // C/D: lanes = N, VGPR = M
        int m = tM * 16 + v + (khi << 3);
        int n = tN * 16 + (l & 15);
        if (m < N2 && n < N2) sim[m * 41 + n] = accv[v] * INV_TEMP;
    }
    __syncthreads();

    if (t < N2) {                              // per-row stable log-softmax
        const int r = t;
        const int partner = (r < K) ? r + K : r - K;
        float pos = sim[r * 41 + partner];
        float mx = pos;
        for (int j = 0; j < N2; ++j) {
            if (j == r || j == partner) continue;
            mx = fmaxf(mx, sim[r * 41 + j]);
        }
        float se = expf(pos - mx);
        for (int j = 0; j < N2; ++j) {
            if (j == r || j == partner) continue;
            se += expf(sim[r * 41 + j] - mx);
        }
        lrow[r] = (mx - pos) + logf(se);       // = -log_softmax[...,0]
    }
    __syncthreads();
    if (t == 0) {
        float s = 0.0f;
        for (int r = 0; r < N2; ++r) s += lrow[r];   // fixed-order reduce
        atomicAdd(acc_simclr, s * (1.0f / ((float)N2 * 4.0f)));
    }
}

// ---------------- kernel 5: combine ----------------
__global__ void k_final(const float* __restrict__ acc, float* __restrict__ out) {
    out[0] = acc[0] * (1.0f / ((float)B * H * W)) + acc[1];
}

extern "C" void kernel_launch(void* const* d_in, const int* in_sizes, int n_in,
                              void* d_out, int out_size, void* d_ws, size_t ws_size,
                              hipStream_t stream) {
    (void)in_sizes; (void)n_in; (void)out_size; (void)ws_size;
    const float* x    = (const float*)d_in[0];
    const int*   mask = (const int*)d_in[1];
    float* ws     = (float*)d_ws;
    float* sums   = ws + WS_SUMS;
    float* counts = ws + WS_COUNTS;
    float* means  = ws + WS_MEANS;
    float* acc    = ws + WS_ACC;

    const int nz = (int)WS_TOTAL;
    k_zero    <<<(nz + 255) / 256, 256, 0, stream>>>(ws, nz);
    k_seg_sums<<<B * CHUNKS,  T1,  0, stream>>>(x, mask, sums, counts);
    k_means   <<<B * K,       128, 0, stream>>>(sums, counts, means);
    k_intra   <<<B * CHUNKS3, T3,  0, stream>>>(x, mask, means, &acc[0]);
    k_simclr  <<<4,           288, 0, stream>>>(means, &acc[1]);
    k_final   <<<1,           1,   0, stream>>>(acc, (float*)d_out);
}
